// ScaledDotProductAttention_57681410785389
// MI455X (gfx1250) — compile-verified, matrix operand order checked by reading, not run on hardware
//
#include <hip/hip_runtime.h>

// Scaled dot-product attention for MI455X (gfx1250, wave32, WMMA + async LDS->global).
// Q,K,V: [B,H,S,D] fp32, mask: [B,H,S,S] bool(u8). Outputs: context [B,H,S,D]
// then attn [B,H,S,S] fp32, concatenated in d_out.
//
// One 16-query tile per 128-thread block (4 waves). Score stripe 16x2048 fp32
// in LDS; softmax in LDS; normalized P stored to HBM via
// global_store_async_from_lds_b128 (immediate-offset burst) so the dominant
// 512MB attn write overlaps the whole P*V WMMA phase. Both GEMMs on
// v_wmma_f32_16x16x32_bf16 with fragments built from contiguous 32B runs per
// the ISA 7.12.2 16-bit layouts.

#define S_LEN 2048
#define D_HEAD 64
#define BH 32            // B*H
#define NTHR 128
#define NWAVE 4
#define KCH 64           // V rows staged per P*V iteration
#define NEGINF (-1e9f)

typedef __attribute__((ext_vector_type(16))) __bf16 v16bf;
typedef __attribute__((ext_vector_type(8)))  __bf16 v8bf;
typedef __attribute__((ext_vector_type(8)))  float  v8f;

static __device__ __forceinline__ unsigned short bfbits(float f) {
  __bf16 b = (__bf16)f;                       // native cvt on gfx1250
  unsigned short s; __builtin_memcpy(&s, &b, 2);
  return s;
}
static __device__ __forceinline__ v16bf cat8(v8bf lo, v8bf hi) {
  return __builtin_shufflevector(lo, hi, 0, 1, 2, 3, 4, 5, 6, 7,
                                         8, 9, 10, 11, 12, 13, 14, 15);
}
static __device__ __forceinline__ v16bf cvt16(v8f lo, v8f hi) {
  v16bf r;
#pragma unroll
  for (int e = 0; e < 8; ++e) { r[e] = (__bf16)lo[e]; r[e + 8] = (__bf16)hi[e]; }
  return r;
}
// Low 32 bits of a generic (flat) shared pointer are the workgroup LDS offset
// (flat->LDS aperture mapping truncates to addr[31:0]).
static __device__ __forceinline__ unsigned lds_off(const void* p) {
  return (unsigned)(size_t)p;
}

// 64 async 16B LDS->global stores per thread; the IOFFSET immediate is added to
// BOTH the global and LDS addresses (ISA 08_async_tensor 4.4), so one base
// address pair + one LDS address VGPR cover the whole stripe.
template <int J>
struct AsyncBurst {
  static __device__ __forceinline__ void run(unsigned long long g, unsigned l) {
    AsyncBurst<J - 1>::run(g, l);
    asm volatile("global_store_async_from_lds_b128 %0, %1, off offset:%2"
                 :: "v"(g), "v"(l), "i"((J - 1) * NTHR * 16) : "memory");
  }
};
template <>
struct AsyncBurst<0> {
  static __device__ __forceinline__ void run(unsigned long long, unsigned) {}
};

__global__ __launch_bounds__(NTHR)
void sdpa_wmma_kernel(const float* __restrict__ Q,
                      const float* __restrict__ K,
                      const float* __restrict__ V,
                      const unsigned char* __restrict__ M,
                      float* __restrict__ ctx,
                      float* __restrict__ attn) {
  __shared__ float sc[16 * S_LEN];             // 128KB: scores -> exp -> normalized P
  __shared__ unsigned short qb[16 * D_HEAD];   // 2KB: Q tile (pre-scaled) bf16
  __shared__ unsigned short vT[D_HEAD * KCH];  // 8KB: V chunk, transposed, bf16
  __shared__ float red[NTHR];                  // row partial reductions
  __shared__ float rowstat[16];                // row max, then 1/sum

  const int tid  = threadIdx.x;
  const int wave = __builtin_amdgcn_readfirstlane(tid >> 5);   // SGPR: uniform
  const int lane = tid & 31;
  const int bh    = blockIdx.x >> 7;           // / (S/16)
  const int qbase = (blockIdx.x & 127) << 4;

  const float* Qb = Q + (size_t)bh * S_LEN * D_HEAD;
  const float* Kb = K + (size_t)bh * S_LEN * D_HEAD;
  const float* Vb = V + (size_t)bh * S_LEN * D_HEAD;
  const unsigned char* Mb = M + (size_t)bh * S_LEN * S_LEN + (size_t)qbase * S_LEN;
  float* attnb = attn + (size_t)bh * S_LEN * S_LEN + (size_t)qbase * S_LEN;
  float* ctxb  = ctx  + (size_t)bh * S_LEN * D_HEAD + (size_t)qbase * D_HEAD;

  // ---- Stage Q tile (16x64, contiguous in memory), folding in 1/sqrt(D) ----
  {
    const float4* Qt = (const float4*)(Qb + (size_t)qbase * D_HEAD);
    ushort4* q4 = (ushort4*)qb;
#pragma unroll
    for (int j = 0; j < 16 * D_HEAD / 4 / NTHR; ++j) {
      const int i = tid + j * NTHR;
      float4 v = Qt[i];
      ushort4 u;
      u.x = bfbits(v.x * 0.125f); u.y = bfbits(v.y * 0.125f);
      u.z = bfbits(v.z * 0.125f); u.w = bfbits(v.w * 0.125f);
      q4[i] = u;
    }
  }
  __syncthreads();

  // Fragment geometry (ISA 7.12.2, 16-bit A/B): elements 0-7 <-> K base..base+7,
  // elements 8-15 <-> K base+16..base+23; base = (lane<16 ? 0 : 8).
  const int mn    = lane & 15;
  const int base  = (lane & 16) ? 8 : 0;
  const int rbase = (lane & 16) ? 8 : 0;       // f32 C/D row base

  v16bf a0, a1;
  {
    const unsigned short* qr = qb + mn * D_HEAD;
    a0 = cat8(*(const v8bf*)(qr + base),      *(const v8bf*)(qr + base + 16));
    a1 = cat8(*(const v8bf*)(qr + 32 + base), *(const v8bf*)(qr + 48 + base));
  }

  // ---- Scores: each wave owns every 4th 16-wide K tile (scalar loop count) ----
  for (int tt = 0; tt < S_LEN / 16 / NWAVE; ++tt) {
    const int t = wave + tt * NWAVE;
    const int col = t * 16 + mn;               // B column = K row index
    const float* Krow = Kb + (size_t)col * D_HEAD;
    __builtin_prefetch(Krow + NWAVE * 16 * D_HEAD, 0, 3);  // speculative, unconditional
    v16bf b0 = cvt16(*(const v8f*)(Krow + base),      *(const v8f*)(Krow + base + 16));
    v16bf b1 = cvt16(*(const v8f*)(Krow + 32 + base), *(const v8f*)(Krow + 48 + base));
    v8f acc = {};
    acc = __builtin_amdgcn_wmma_f32_16x16x32_bf16(false, a0, false, b0,
                                                  (short)0, acc, false, false);
    acc = __builtin_amdgcn_wmma_f32_16x16x32_bf16(false, a1, false, b1,
                                                  (short)0, acc, false, false);
#pragma unroll
    for (int r = 0; r < 8; ++r)
      sc[(rbase + r) * S_LEN + col] = acc[r];
  }
  __syncthreads();

  // ---- Softmax pass 1: apply mask (coalesced uint4 = 16 bytes) + row max ----
  {
    const int r = tid >> 3, l8 = tid & 7;      // 8 threads per row
    float* sr = sc + r * S_LEN;
    const uint4* mr = (const uint4*)(Mb + (size_t)r * S_LEN);
    float mx = -3.0e38f;
    for (int j = 0; j < S_LEN / 16 / 8; ++j) { // 16 iterations, scalar count
      const int g = l8 + j * 8;
      uint4 m16 = mr[g];
      unsigned mw[4] = {m16.x, m16.y, m16.z, m16.w};
      float4* sp = (float4*)(sr + g * 16);
#pragma unroll
      for (int q = 0; q < 4; ++q) {
        float4 s4 = sp[q];
        unsigned m = mw[q];
        if (m & 0x000000ffu) s4.x = NEGINF;
        if (m & 0x0000ff00u) s4.y = NEGINF;
        if (m & 0x00ff0000u) s4.z = NEGINF;
        if (m & 0xff000000u) s4.w = NEGINF;
        sp[q] = s4;
        mx = fmaxf(mx, fmaxf(fmaxf(s4.x, s4.y), fmaxf(s4.z, s4.w)));
      }
    }
    red[tid] = mx;
  }
  __syncthreads();
  if (tid < 16) {
    float mx = red[tid * 8];
#pragma unroll
    for (int j = 1; j < 8; ++j) mx = fmaxf(mx, red[tid * 8 + j]);
    rowstat[tid] = mx;
  }
  __syncthreads();

  // ---- Softmax pass 2: exp in place + row sum ----
  {
    const int r = tid >> 3, l8 = tid & 7;
    const float mx = rowstat[r];
    float* sr = sc + r * S_LEN;
    float sum = 0.f;
    for (int j = 0; j < S_LEN / 4 / 8; ++j) {  // 64 iterations, scalar count
      const int g = l8 + j * 8;
      float4 s4 = *(float4*)(sr + g * 4);
      s4.x = __expf(s4.x - mx); s4.y = __expf(s4.y - mx);
      s4.z = __expf(s4.z - mx); s4.w = __expf(s4.w - mx);
      *(float4*)(sr + g * 4) = s4;
      sum += (s4.x + s4.y) + (s4.z + s4.w);
    }
    red[tid] = sum;
  }
  __syncthreads();
  if (tid < 16) {
    float sum = 0.f;
#pragma unroll
    for (int j = 0; j < 8; ++j) sum += red[tid * 8 + j];
    rowstat[tid] = 1.0f / sum;
  }
  __syncthreads();

  // ---- Normalize P in place in LDS ----
  {
    float4* s4p = (float4*)sc;
    for (int j = 0; j < 16 * S_LEN / 4 / NTHR; ++j) {  // 64 iterations
      const int i = tid + j * NTHR;
      const float inv = rowstat[i >> 9];
      float4 s4 = s4p[i];
      s4.x *= inv; s4.y *= inv; s4.z *= inv; s4.w *= inv;
      s4p[i] = s4;
    }
  }
  __syncthreads();

  // ---- Fire attn stripe to HBM: async LDS->global, overlaps all of P*V ----
  // (sc is only READ below; ASYNCcnt drains under S_ENDPGM's implicit wait-idle.)
  {
    const unsigned long long g =
        (unsigned long long)(size_t)(attnb + (size_t)tid * 4);
    const unsigned l = lds_off(sc) + (unsigned)tid * 16u;
    AsyncBurst<16 * S_LEN / 4 / NTHR>::run(g, l);      // 64 stores, offset imm
  }

  // ---- context = P(16x2048) @ V(2048x64); each wave one 16-col tile ----
  {
    const int n0 = wave * 16;
    v8f acc = {};
    for (int kk0 = 0; kk0 < S_LEN; kk0 += KCH) {
      __syncthreads();                          // vT safe to overwrite
      // Stage V[kk0..kk0+KCH) x 64 into LDS transposed (bf16), 4x4 blocks.
      {
        const float* Vsrc = Vb + (size_t)kk0 * D_HEAD;
#pragma unroll
        for (int it = 0; it < (KCH / 4) * (D_HEAD / 4) / NTHR; ++it) {
          const int blk = tid + it * NTHR;
          const int bk = blk >> 4, bc = blk & 15;
          __builtin_prefetch(Vsrc + KCH * D_HEAD + (size_t)(bk * 4) * D_HEAD + bc * 4, 0, 3);
          float a[4][4];
#pragma unroll
          for (int rr = 0; rr < 4; ++rr) {
            float4 ld = *(const float4*)(Vsrc + (size_t)(bk * 4 + rr) * D_HEAD + bc * 4);
            a[rr][0] = ld.x; a[rr][1] = ld.y; a[rr][2] = ld.z; a[rr][3] = ld.w;
          }
#pragma unroll
          for (int cc = 0; cc < 4; ++cc) {
            ushort4 u;
            u.x = bfbits(a[0][cc]); u.y = bfbits(a[1][cc]);
            u.z = bfbits(a[2][cc]); u.w = bfbits(a[3][cc]);
            *(ushort4*)&vT[(bc * 4 + cc) * KCH + bk * 4] = u;
          }
        }
      }
      __syncthreads();
#pragma unroll
      for (int half = 0; half < 2; ++half) {
        const int kk = kk0 + half * 32;
        const float* pr = sc + mn * S_LEN + kk;      // normalized P row
        v16bf pa = cvt16(*(const v8f*)(pr + base), *(const v8f*)(pr + base + 16));
        const unsigned short* vr = vT + (n0 + mn) * KCH + half * 32;
        v16bf vb = cat8(*(const v8bf*)(vr + base), *(const v8bf*)(vr + base + 16));
        acc = __builtin_amdgcn_wmma_f32_16x16x32_bf16(false, pa, false, vb,
                                                      (short)0, acc, false, false);
      }
    }
#pragma unroll
    for (int r = 0; r < 8; ++r) {
      const int mrow = rbase + r;
      ctxb[(size_t)mrow * D_HEAD + n0 + mn] = acc[r];
    }
  }

  // Explicit drain of the async attn stores (also implied by S_ENDPGM wait-idle).
  asm volatile("s_wait_asynccnt 0x0" ::: "memory");
}

extern "C" void kernel_launch(void* const* d_in, const int* in_sizes, int n_in,
                              void* d_out, int out_size, void* d_ws, size_t ws_size,
                              hipStream_t stream) {
  (void)in_sizes; (void)n_in; (void)out_size; (void)d_ws; (void)ws_size;
  const float* Q = (const float*)d_in[0];
  const float* K = (const float*)d_in[1];
  const float* V = (const float*)d_in[2];
  const unsigned char* M = (const unsigned char*)d_in[3];
  float* ctx  = (float*)d_out;
  float* attn = ctx + (size_t)BH * S_LEN * D_HEAD;   // context first, then attn
  dim3 grid(BH * (S_LEN / 16));                      // 4096 blocks
  sdpa_wmma_kernel<<<grid, NTHR, 0, stream>>>(Q, K, V, M, ctx, attn);
}
